// CustomGNN_34050500722941
// MI455X (gfx1250) — matrix operand order repaired and checked by reading, not compiled, not run on hardware
//
#include <hip/hip_runtime.h>

#define N_NODES 100000
#define N_EDGES 1600000
#define HIDDEN  64

typedef __attribute__((ext_vector_type(16))) _Float16 v16h;
typedef __attribute__((ext_vector_type(8)))  float    v8f;

// ---------------------------------------------------------------------------
// A-matrix (16x32 f16) per-lane load from an f32 row. Lane L: M = L&15,
// half = L>>4. Element s holds K = 16*half+s (contiguous, matching B).
// koff selects the 32-wide K slice of the 64-wide row.
// ---------------------------------------------------------------------------
__device__ __forceinline__ v16h load_a16(const float* __restrict__ row, int half, int koff) {
  v16h a;
#pragma unroll
  for (int i = 0; i < 8; ++i) {
    a[i]     = (_Float16)row[koff + 16 * half + i];
    a[i + 8] = (_Float16)row[koff + 16 * half + 8 + i];
  }
  return a;
}

// ---------------------------------------------------------------------------
// Weight f32 -> f16 pre-conversion (once per launch; 64x64 per matrix).
// ---------------------------------------------------------------------------
__global__ void __launch_bounds__(256) gnn_cvt_w(const float* __restrict__ w,
                                                 _Float16* __restrict__ o, int n) {
  int i = blockIdx.x * 256 + threadIdx.x;
  if (i < n) o[i] = (_Float16)w[i];
}

// ---------------------------------------------------------------------------
// Pre-MP dense: h = relu(x @ preW.T + preb), IN_DIM = 3
// ---------------------------------------------------------------------------
__global__ void __launch_bounds__(256) gnn_pre(const float* __restrict__ x,
                                               const float* __restrict__ preW,
                                               const float* __restrict__ preb,
                                               float* __restrict__ h) {
  int idx = blockIdx.x * 256 + threadIdx.x;
  if (idx >= N_NODES * HIDDEN) return;
  int i = idx >> 6;
  int j = idx & 63;
  const float* xr = x + (size_t)i * 3;
  const float* w  = preW + j * 3;
  float v = preb[j] + w[0] * xr[0] + w[1] * xr[1] + w[2] * xr[2];
  h[idx] = fmaxf(v, 0.0f);
}

// ---------------------------------------------------------------------------
// Edge scatter: agg[dst] += h[src].
// Block = 256 threads = 8 waves; 2 waves per edge (32 features each).
// The edge id is wave-uniform -> readfirstlane makes src/dst loads scalar
// (s_load), eliminating 64x-redundant index traffic. Gathers of h[src] are
// 128B-coalesced per wave; atomics hit the L2-resident agg buffer.
// ---------------------------------------------------------------------------
__global__ void __launch_bounds__(256) gnn_scatter(const float* __restrict__ h,
                                                   const long long* __restrict__ src,
                                                   const long long* __restrict__ dst,
                                                   float* __restrict__ agg) {
  int wave = __builtin_amdgcn_readfirstlane(threadIdx.x >> 5);  // 0..7, wave-uniform
  int lane = threadIdx.x & 31;
  long long e = (long long)blockIdx.x * 4 + (wave >> 1);        // scalar
  if (e >= N_EDGES) return;
  int f = (wave & 1) * 32 + lane;                               // 0..63
  int s = (int)src[e];                                          // scalar load
  int d = (int)dst[e];                                          // scalar load
  atomicAdd(agg + (size_t)d * HIDDEN + f, h[(size_t)s * HIDDEN + f]);
}

// ---------------------------------------------------------------------------
// GraphConv dense part via WMMA:
//   out = relu(agg @ relW.T + relb + h @ rootW.T)
// Block = 256 threads = 8 waves; block covers 32 nodes.
// Wave w: row-tile rt = w>>2 (16 nodes), col-tile nt = w&3 (16 features).
// B tiles are pre-converted f16 (direct b128 loads); A tiles convert f32->f16
// in-register. 4x v_wmma_f32_16x16x32_f16 accumulate one f32 16x16 tile.
// EXEC is all-ones (exact grid; no branches) as WMMA requires.
// ---------------------------------------------------------------------------
__global__ void __launch_bounds__(256) gnn_mp_gemm(const float* __restrict__ agg,
                                                   const float* __restrict__ h,
                                                   const _Float16* __restrict__ relW16,
                                                   const float* __restrict__ relb,
                                                   const _Float16* __restrict__ rootW16,
                                                   float* __restrict__ out) {
  const int tid  = threadIdx.x;
  const int lane = tid & 31;
  const int wave = tid >> 5;
  const int rt   = wave >> 2;   // 0..1
  const int nt   = wave & 3;    // 0..3
  const int half = lane >> 4;   // 0..1
  const int lid  = lane & 15;

  const int rowbase = blockIdx.x * 32 + rt * 16;
  const int jcol    = nt * 16 + lid;

  const float*    aggrow  = agg     + (size_t)(rowbase + lid) * HIDDEN;
  const float*    hrow    = h       + (size_t)(rowbase + lid) * HIDDEN;
  const _Float16* relrow  = relW16  + (size_t)jcol * HIDDEN;
  const _Float16* rootrow = rootW16 + (size_t)jcol * HIDDEN;

  // B layout (32x16 f16): lane holds column N = lid, K = 16*half .. 16*half+15
  // contiguous -> a single 32-byte aligned v16h load per K-slice.
  v16h b0 = *(const v16h*)(relrow  + 16 * half);
  v16h b1 = *(const v16h*)(relrow  + 32 + 16 * half);
  v16h b2 = *(const v16h*)(rootrow + 16 * half);
  v16h b3 = *(const v16h*)(rootrow + 32 + 16 * half);

  v16h a0 = load_a16(aggrow, half, 0);
  v16h a1 = load_a16(aggrow, half, 32);
  v16h a2 = load_a16(hrow,   half, 0);
  v16h a3 = load_a16(hrow,   half, 32);

  v8f c = {};
  c = __builtin_amdgcn_wmma_f32_16x16x32_f16(false, a0, false, b0, (short)0, c, false, false);
  c = __builtin_amdgcn_wmma_f32_16x16x32_f16(false, a1, false, b1, (short)0, c, false, false);
  c = __builtin_amdgcn_wmma_f32_16x16x32_f16(false, a2, false, b2, (short)0, c, false, false);
  c = __builtin_amdgcn_wmma_f32_16x16x32_f16(false, a3, false, b3, (short)0, c, false, false);

  const float bias = relb[jcol];
#pragma unroll
  for (int v = 0; v < 8; ++v) {
    int m = rowbase + v + 8 * half;           // C/D layout: lane N=L&15, VGPR v -> M=v+8*half
    out[(size_t)m * HIDDEN + jcol] = fmaxf(c[v] + bias, 0.0f);
  }
}

// ---------------------------------------------------------------------------
// Post-MP dense: out = relu(h @ postW.T + postb), OUT_DIM = 2
// ---------------------------------------------------------------------------
__global__ void __launch_bounds__(128) gnn_post(const float* __restrict__ h,
                                                const float* __restrict__ postW,
                                                const float* __restrict__ postb,
                                                float* __restrict__ out) {
  int i = blockIdx.x * 128 + threadIdx.x;
  if (i >= N_NODES) return;
  const float4* hr = (const float4*)(h + (size_t)i * HIDDEN);
  const float4* w0 = (const float4*)(postW);
  const float4* w1 = (const float4*)(postW + HIDDEN);
  float acc0 = postb[0], acc1 = postb[1];
#pragma unroll
  for (int k = 0; k < 16; ++k) {
    float4 hv = hr[k], a = w0[k], b = w1[k];
    acc0 += hv.x * a.x + hv.y * a.y + hv.z * a.z + hv.w * a.w;
    acc1 += hv.x * b.x + hv.y * b.y + hv.z * b.z + hv.w * b.w;
  }
  out[(size_t)i * 2]     = fmaxf(acc0, 0.0f);
  out[(size_t)i * 2 + 1] = fmaxf(acc1, 0.0f);
}

extern "C" void kernel_launch(void* const* d_in, const int* in_sizes, int n_in,
                              void* d_out, int out_size, void* d_ws, size_t ws_size,
                              hipStream_t stream) {
  const float*     x     = (const float*)d_in[0];
  const long long* ei    = (const long long*)d_in[1];
  const float*     preW  = (const float*)d_in[2];
  const float*     preb  = (const float*)d_in[3];
  const float*     postW = (const float*)d_in[4];
  const float*     postb = (const float*)d_in[5];
  const float* relW[3]  = {(const float*)d_in[6], (const float*)d_in[9],  (const float*)d_in[12]};
  const float* relb[3]  = {(const float*)d_in[7], (const float*)d_in[10], (const float*)d_in[13]};
  const float* rootW[3] = {(const float*)d_in[8], (const float*)d_in[11], (const float*)d_in[14]};

  const long long* src = ei;
  const long long* dst = ei + N_EDGES;

  const size_t feat_bytes = (size_t)N_NODES * HIDDEN * sizeof(float);
  float* b0 = (float*)d_ws;
  float* b1 = (float*)((char*)d_ws + feat_bytes);
  float* b2 = (float*)((char*)d_ws + 2 * feat_bytes);

  // f16 weight staging area (6 x 64x64 halves = 48 KB), 32B-aligned.
  _Float16* w16base = (_Float16*)((char*)d_ws + 3 * feat_bytes);
  _Float16* relW16[3]  = {w16base,            w16base + 1 * 4096, w16base + 2 * 4096};
  _Float16* rootW16[3] = {w16base + 3 * 4096, w16base + 4 * 4096, w16base + 5 * 4096};

  const int WN = HIDDEN * HIDDEN;  // 4096
  for (int l = 0; l < 3; ++l) {
    gnn_cvt_w<<<(WN + 255) / 256, 256, 0, stream>>>(relW[l],  relW16[l],  WN);
    gnn_cvt_w<<<(WN + 255) / 256, 256, 0, stream>>>(rootW[l], rootW16[l], WN);
  }

  // Pre-MP: x -> b0
  gnn_pre<<<(N_NODES * HIDDEN + 255) / 256, 256, 0, stream>>>(x, preW, preb, b0);

  float* hbuf = b0;  // current activations
  float* abuf = b1;  // aggregation scratch
  float* obuf = b2;  // next activations
  for (int l = 0; l < 3; ++l) {
    hipMemsetAsync(abuf, 0, feat_bytes, stream);
    gnn_scatter<<<(N_EDGES + 3) / 4, 256, 0, stream>>>(hbuf, src, dst, abuf);
    gnn_mp_gemm<<<N_NODES / 32, 256, 0, stream>>>(abuf, hbuf, relW16[l], relb[l],
                                                  rootW16[l], obuf);
    float* t = hbuf; hbuf = obuf; obuf = abuf; abuf = t;     // rotate buffers
  }

  gnn_post<<<(N_NODES + 127) / 128, 128, 0, stream>>>(hbuf, postW, postb, (float*)d_out);
}
